// GCN_44092134260958
// MI455X (gfx1250) — compile-verified, matrix operand order checked by reading, not run on hardware
//
#include <hip/hip_runtime.h>
#include <hip/hip_bf16.h>

#define N_NODES    100000
#define N_EDGES    3200000
#define IN_DIM     128
#define EMB        32
#define NUM_GRAPHS 256
#define M_TILES    (N_NODES / 16)   // 6250 exactly

typedef __attribute__((ext_vector_type(16))) __bf16 v16bf;
typedef __attribute__((ext_vector_type(8)))  float  v8f;

// ---------------------------------------------------------------------------
// 0) zero scratch (agg, deg, graph sums/counts) -- ws is poisoned by harness
// ---------------------------------------------------------------------------
__global__ void gcn_zero_ws(float* __restrict__ agg, unsigned* __restrict__ deg,
                            float* __restrict__ gsum, float* __restrict__ gcnt) {
    int i = blockIdx.x * blockDim.x + threadIdx.x;
    int stride = gridDim.x * blockDim.x;
    for (int j = i; j < N_NODES * EMB; j += stride) agg[j] = 0.0f;
    for (int j = i; j < N_NODES; j += stride)       deg[j] = 0u;
    for (int j = i; j < NUM_GRAPHS * EMB; j += stride) gsum[j] = 0.0f;
    for (int j = i; j < NUM_GRAPHS; j += stride)       gcnt[j] = 0.0f;
}

// ---------------------------------------------------------------------------
// 1) xw = x @ W1 via v_wmma_f32_16x16x32_bf16. One wave per 16-row tile,
//    two 16x16 accumulators (EMB=32), 4 K-steps of 32.
//    W1 staged in LDS transposed (Wt[n][k], bf16) so B-fragments are
//    contiguous 8-element runs per the CDNA5 16-bit operand layout.
// ---------------------------------------------------------------------------
__global__ __launch_bounds__(256)
void gcn_xw_wmma(const float* __restrict__ x, const float* __restrict__ W,
                 float* __restrict__ xw) {
    __shared__ __bf16 Wt[EMB * IN_DIM];          // 8 KB of 320 KB WGP LDS
    for (int idx = threadIdx.x; idx < IN_DIM * EMB; idx += blockDim.x) {
        int k = idx / EMB, n = idx % EMB;        // W is [K=128][N=32] row-major
        Wt[n * IN_DIM + k] = (__bf16)W[idx];
    }
    __syncthreads();

    int wave = blockIdx.x * (blockDim.x >> 5) + (threadIdx.x >> 5);
    if (wave >= M_TILES) return;
    int lane = threadIdx.x & 31;
    int half = lane >> 4;                        // lanes 16-31: +8 in K pattern / +8 in M for C/D
    int r    = lane & 15;

    // Preload all 8 B fragments (4 K-steps x 2 N-tiles) from LDS.
    v16bf bfrag[4][2];
#pragma unroll
    for (int ks = 0; ks < 4; ++ks)
#pragma unroll
        for (int nt = 0; nt < 2; ++nt) {
            int n = nt * 16 + r;
            const __bf16* wrow = &Wt[n * IN_DIM + ks * 32 + half * 8];
#pragma unroll
            for (int j = 0; j < 8; ++j) {        // elems 0..7: K=half*8+j ; 8..15: K=16+half*8+j
                bfrag[ks][nt][j]     = wrow[j];
                bfrag[ks][nt][8 + j] = wrow[16 + j];
            }
        }

    v8f acc0 = {}, acc1 = {};
    const float* xrow = x + (size_t)(wave * 16 + r) * IN_DIM;
#pragma unroll
    for (int ks = 0; ks < 4; ++ks) {
        v16bf a;
        int k0 = ks * 32 + half * 8;
#pragma unroll
        for (int j = 0; j < 8; ++j) {            // two contiguous 32B runs per lane
            a[j]     = (__bf16)xrow[k0 + j];
            a[8 + j] = (__bf16)xrow[k0 + 16 + j];
        }
        acc0 = __builtin_amdgcn_wmma_f32_16x16x32_bf16(false, a, false, bfrag[ks][0],
                                                       (short)0, acc0, false, false);
        acc1 = __builtin_amdgcn_wmma_f32_16x16x32_bf16(false, a, false, bfrag[ks][1],
                                                       (short)0, acc1, false, false);
    }

    // C/D layout: VGPR i -> M = i + half*8, N = r (+16 for second tile)
    int base = wave * 16;
#pragma unroll
    for (int i = 0; i < 8; ++i) {
        size_t orow = (size_t)(base + i + half * 8) * EMB;
        xw[orow + r]      = acc0[i];
        xw[orow + 16 + r] = acc1[i];
    }
}

// ---------------------------------------------------------------------------
// 2) in-degree over dst (self-loop handled as +1 in rsqrt)
// ---------------------------------------------------------------------------
__global__ void gcn_degree(const int* __restrict__ dst, unsigned* __restrict__ deg) {
    int e = blockIdx.x * blockDim.x + threadIdx.x;
    if (e < N_EDGES) atomicAdd(&deg[dst[e]], 1u);
}

__global__ void gcn_dinv(const unsigned* __restrict__ deg, float* __restrict__ dinv) {
    int i = blockIdx.x * blockDim.x + threadIdx.x;
    if (i < N_NODES) dinv[i] = rsqrtf((float)deg[i] + 1.0f);
}

// ---------------------------------------------------------------------------
// 3) message scatter: agg[dst] += xw[src] * dinv[src]*dinv[dst].
//    xw (12.8MB) + agg (12.8MB) are L2-resident (192MB); only the edge
//    stream hits HBM -> prefetch it ahead.
// ---------------------------------------------------------------------------
__global__ void gcn_edges(const int* __restrict__ src, const int* __restrict__ dst,
                          const float* __restrict__ xw, const float* __restrict__ dinv,
                          float* __restrict__ agg) {
    int e = blockIdx.x * blockDim.x + threadIdx.x;
    if (e >= N_EDGES) return;
    if (e + 8192 < N_EDGES) {
        __builtin_prefetch(&src[e + 8192], 0, 1);
        __builtin_prefetch(&dst[e + 8192], 0, 1);
    }
    int s = src[e], d = dst[e];
    float w = dinv[s] * dinv[d];
    const float4* xr = (const float4*)(xw + (size_t)s * EMB);
    float* ar = agg + (size_t)d * EMB;
#pragma unroll
    for (int q = 0; q < 8; ++q) {
        float4 v = xr[q];
        atomicAdd(&ar[q * 4 + 0], v.x * w);
        atomicAdd(&ar[q * 4 + 1], v.y * w);
        atomicAdd(&ar[q * 4 + 2], v.z * w);
        atomicAdd(&ar[q * 4 + 3], v.w * w);
    }
}

// ---------------------------------------------------------------------------
// 4) h = tanh(agg + xw*dinv^2 + b1); pool into per-graph sums/counts
// ---------------------------------------------------------------------------
__global__ void gcn_node(const float* __restrict__ xw, const float* __restrict__ agg,
                         const float* __restrict__ dinv, const float* __restrict__ b1,
                         const int* __restrict__ batch,
                         float* __restrict__ gsum, float* __restrict__ gcnt) {
    int i = blockIdx.x * blockDim.x + threadIdx.x;
    if (i >= N_NODES * EMB) return;
    int node = i / EMB, c = i % EMB;
    float di = dinv[node];
    float h = tanhf(agg[i] + xw[i] * di * di + b1[c]);
    int g = batch[node];
    atomicAdd(&gsum[g * EMB + c], h);
    if (c == 0) atomicAdd(&gcnt[g], 1.0f);
}

// ---------------------------------------------------------------------------
// 5) out[g] = [sum, mean] @ Wout + bout  (256 outputs; tiny GEMV)
// ---------------------------------------------------------------------------
__global__ void gcn_out(const float* __restrict__ gsum, const float* __restrict__ gcnt,
                        const float* __restrict__ Wout, const float* __restrict__ bout,
                        float* __restrict__ out) {
    int g = blockIdx.x * blockDim.x + threadIdx.x;
    if (g >= NUM_GRAPHS) return;
    float inv_cnt = 1.0f / fmaxf(gcnt[g], 1.0f);
    float acc = bout[0];
#pragma unroll
    for (int c = 0; c < EMB; ++c) {
        float s = gsum[g * EMB + c];
        acc += s * Wout[c] + (s * inv_cnt) * Wout[EMB + c];
    }
    out[g] = acc;
}

// ---------------------------------------------------------------------------
extern "C" void kernel_launch(void* const* d_in, const int* in_sizes, int n_in,
                              void* d_out, int out_size, void* d_ws, size_t ws_size,
                              hipStream_t stream) {
    const float* x     = (const float*)d_in[0];
    const int*   eidx  = (const int*)d_in[1];
    const int*   batch = (const int*)d_in[2];
    const float* W1    = (const float*)d_in[3];
    const float* b1    = (const float*)d_in[4];
    const float* Wout  = (const float*)d_in[5];
    const float* bout  = (const float*)d_in[6];
    float* out = (float*)d_out;

    const int* src = eidx;             // edge_index[0]
    const int* dst = eidx + N_EDGES;   // edge_index[1]

    char* ws = (char*)d_ws;
    float*    xw   = (float*)(ws);                                 // 12.8 MB
    float*    agg  = (float*)(ws + 12800000);                      // 12.8 MB
    unsigned* deg  = (unsigned*)(ws + 25600000);                   // 0.4 MB
    float*    dinv = (float*)(ws + 26000000);                      // 0.4 MB
    float*    gsum = (float*)(ws + 26400000);                      // 32 KB
    float*    gcnt = (float*)(ws + 26432768);                      // 1 KB

    gcn_zero_ws<<<1024, 256, 0, stream>>>(agg, deg, gsum, gcnt);

    // 6250 tiles, 8 waves per 256-thread block
    gcn_xw_wmma<<<(M_TILES + 7) / 8, 256, 0, stream>>>(x, W1, xw);

    gcn_degree<<<(N_EDGES + 255) / 256, 256, 0, stream>>>(dst, deg);
    gcn_dinv<<<(N_NODES + 255) / 256, 256, 0, stream>>>(deg, dinv);

    gcn_edges<<<(N_EDGES + 255) / 256, 256, 0, stream>>>(src, dst, xw, dinv, agg);

    gcn_node<<<(N_NODES * EMB + 255) / 256, 256, 0, stream>>>(xw, agg, dinv, b1,
                                                              batch, gsum, gcnt);

    gcn_out<<<1, 256, 0, stream>>>(gsum, gcnt, Wout, bout, out);
}
